// FastPropagation_82205674045866
// MI455X (gfx1250) — matrix operand order repaired and checked by reading, not compiled, not run on hardware
//
#include <hip/hip_runtime.h>
#include <hip/hip_bf16.h>

typedef __attribute__((ext_vector_type(16))) _Float16 v16h;
typedef __attribute__((ext_vector_type(8)))  _Float16 v8h;
typedef __attribute__((ext_vector_type(8)))  float    v8f;

#define H_IMG 512
#define W_IMG 640
#define TH 32
#define TW 32
// h1 tile: (TH+2) x (TW+2) = 34x34 ; input tile: 36x36
#define H1H (TH + 2)
#define H1W (TW + 2)
#define INH (TH + 4)
#define INW (TW + 4)
#define NPIX_H1 (H1H * H1W)           /* 1156 */
#define NT_H1   ((NPIX_H1 + 15) / 16) /* 73 N-tiles for GEMM1 */

static __device__ __forceinline__ v16h cat16(v8h lo, v8h hi) {
    v16h r;
#pragma unroll
    for (int i = 0; i < 8; ++i) { r[i] = lo[i]; r[i + 8] = hi[i]; }
    return r;
}

__launch_bounds__(256, 1)
__global__ void FastPropagation_82205674045866_kernel(
        const float* __restrict__ depth,  // [B,1,H,W]
        const float* __restrict__ img,    // [B,16,H,W]
        const float* __restrict__ w1,     // [32,16,3,3]
        const float* __restrict__ b1,     // [32]
        const float* __restrict__ gamma,  // [32]
        const float* __restrict__ beta,   // [32]
        const float* __restrict__ rmean,  // [32]
        const float* __restrict__ rvar,   // [32]
        const float* __restrict__ w2,     // [9,32,3,3]
        float* __restrict__ out)          // [B,H,W]
{
    // f16 input halo tile, [y][x][c] (c contiguous -> 16B/32B LDS fragments)
    __shared__ __align__(32) _Float16 sIn[INH * INW * 16];       // 41472 B
    // conv1 weights [tap(10, tap9=0)][oc 32][ic 16]
    __shared__ __align__(16) _Float16 sWa[10 * 32 * 16];         // 10240 B
    // conv2 weights [tap 9][oc 16 (9 valid)][ic 32]
    __shared__ __align__(16) _Float16 sWb[9 * 16 * 32];          //  9216 B
    // h1 tile, [pixel (padded to 73*16)][oc 32]
    __shared__ __align__(32) _Float16 sH1[NT_H1 * 16 * 32];      // 74752 B
    __shared__ float sDepth[H1H * H1W];                          //  4624 B
    __shared__ __align__(32) float sScale[32];
    __shared__ __align__(32) float sShift[32];

    const int tid  = threadIdx.x;
    const int lane = tid & 31;
    const int wid  = tid >> 5;
    const int half = lane >> 4;   // 0 | 1
    const int lr   = lane & 15;

    const int gx0 = blockIdx.x * TW;
    const int gy0 = blockIdx.y * TH;
    const int b   = blockIdx.z;

    // ---------------- Phase 0: stage everything into LDS ----------------
    if (tid < 32) {
        float s = gamma[tid] * rsqrtf(rvar[tid] + 1e-5f);
        sScale[tid] = s;
        sShift[tid] = b1[tid] * s + beta[tid] - rmean[tid] * s;
    }
    // conv1 weights -> f16, padded to 10 taps
    for (int idx = tid; idx < 10 * 32 * 16; idx += 256) {
        const int tap = idx >> 9;          // /512
        const int rem = idx & 511;
        const int oc  = rem >> 4;
        const int ic  = rem & 15;
        float v = (tap < 9) ? w1[(oc * 16 + ic) * 9 + tap] : 0.0f;
        sWa[idx] = (_Float16)v;
    }
    // conv2 weights -> f16, oc padded 9..15 = 0
    for (int idx = tid; idx < 9 * 16 * 32; idx += 256) {
        const int tap = idx >> 9;          // /512
        const int rem = idx & 511;
        const int oc  = rem >> 5;
        const int ic  = rem & 31;
        float v = (oc < 9) ? w2[(oc * 32 + ic) * 9 + tap] : 0.0f;
        sWb[idx] = (_Float16)v;
    }
    // depth halo tile with global replicate clamp
    {
        const float* dpb = depth + (size_t)b * H_IMG * W_IMG;
        for (int idx = tid; idx < NPIX_H1; idx += 256) {
            const int i  = idx / H1W;
            const int j  = idx % H1W;
            int gy = gy0 - 1 + i; gy = gy < 0 ? 0 : (gy > H_IMG - 1 ? H_IMG - 1 : gy);
            int gx = gx0 - 1 + j; gx = gx < 0 ? 0 : (gx > W_IMG - 1 ? W_IMG - 1 : gx);
            sDepth[idx] = dpb[gy * W_IMG + gx];
        }
    }
    // input halo tile (zero conv padding), coalesced over x
    {
        const float* imb = img + (size_t)b * 16 * H_IMG * W_IMG;
        for (int idx = tid; idx < 16 * INH * INW; idx += 256) {
            const int c   = idx / (INH * INW);
            const int rem = idx % (INH * INW);
            const int y   = rem / INW;
            const int x   = rem % INW;
            const int gy  = gy0 - 2 + y;
            const int gx  = gx0 - 2 + x;
            float v = 0.0f;
            if (gy >= 0 && gy < H_IMG && gx >= 0 && gx < W_IMG)
                v = imb[(size_t)c * H_IMG * W_IMG + gy * W_IMG + gx];
            sIn[(y * INW + x) * 16 + c] = (_Float16)v;
        }
    }
    __syncthreads();

    // ---------------- Phase 1: conv1 + BN + ReLU (implicit GEMM) --------
    // M = 32 oc (2 tiles, shared B), N = 1156 h1 pixels (73 tiles),
    // K = 144 as 5 steps of 32 (two taps / step, phantom tap9 weight = 0)
    {
        // loop-invariant A fragments: 5 k-steps x 2 M-tiles
        v16h A0[5], A1[5];
#pragma unroll
        for (int s = 0; s < 5; ++s) {
            const int tap0 = 2 * s, tap1 = 2 * s + 1;
            const int tapa = half ? tap1 : tap0;
            A0[s] = cat16(*(const v8h*)&sWa[(tapa * 32 + lr) * 16],
                          *(const v8h*)&sWa[(tapa * 32 + lr) * 16 + 8]);
            A1[s] = cat16(*(const v8h*)&sWa[(tapa * 32 + 16 + lr) * 16],
                          *(const v8h*)&sWa[(tapa * 32 + 16 + lr) * 16 + 8]);
        }
        const v8f scl0 = *(const v8f*)&sScale[half * 8];
        const v8f scl1 = *(const v8f*)&sScale[16 + half * 8];
        const v8f sft0 = *(const v8f*)&sShift[half * 8];
        const v8f sft1 = *(const v8f*)&sShift[16 + half * 8];

        for (int nt = wid; nt < NT_H1; nt += 8) {
            const int p  = nt * 16 + lr;
            const int pc = p < NPIX_H1 ? p : (NPIX_H1 - 1);
            const int y1 = pc / H1W;
            const int x1 = pc % H1W;
            v8f acc0 = {};
            v8f acc1 = {};
#pragma unroll
            for (int s = 0; s < 5; ++s) {
                // B fragment: lane-half selects K block of 16 => one tap per half
                const int tapb = (2 * s + half) > 8 ? 8 : (2 * s + half);
                const int dy   = tapb / 3;
                const int dx   = tapb % 3;
                const v16h bb  = *(const v16h*)&sIn[((y1 + dy) * INW + (x1 + dx)) * 16];
                acc0 = __builtin_amdgcn_wmma_f32_16x16x32_f16(
                        false, A0[s], false, bb, (short)0, acc0, false, false);
                acc1 = __builtin_amdgcn_wmma_f32_16x16x32_f16(
                        false, A1[s], false, bb, (short)0, acc1, false, false);
            }
            v8h hv0, hv1;
#pragma unroll
            for (int i = 0; i < 8; ++i) {
                float v0 = fmaxf(acc0[i] * scl0[i] + sft0[i], 0.0f);
                float v1 = fmaxf(acc1[i] * scl1[i] + sft1[i], 0.0f);
                hv0[i] = (_Float16)v0;
                hv1[i] = (_Float16)v1;
            }
            // unconditional store: pad pixels (p >= 1156) are never read back
            *(v8h*)&sH1[p * 32 + half * 8]      = hv0;
            *(v8h*)&sH1[p * 32 + 16 + half * 8] = hv1;
        }
    }
    __syncthreads();

    // ---------------- Phase 2: conv2 + softmax + depth gather -----------
    // M = 16 (9 valid oc), N = 1024 out pixels (64 tiles), K = 288 (9 taps x 32)
    {
        v16h A2[9];
#pragma unroll
        for (int tap = 0; tap < 9; ++tap)
            A2[tap] = cat16(*(const v8h*)&sWb[(tap * 16 + lr) * 32 + half * 16],
                            *(const v8h*)&sWb[(tap * 16 + lr) * 32 + half * 16 + 8]);

        const size_t out_base = ((size_t)b * H_IMG + gy0) * W_IMG + gx0;
        for (int nt = wid; nt < 64; nt += 8) {
            const int p = nt * 16 + lr;
            const int y = p >> 5;
            const int x = p & 31;
            v8f acc = {};
#pragma unroll
            for (int tap = 0; tap < 9; ++tap) {
                const int dy = tap / 3;
                const int dx = tap % 3;
                const v16h bb = *(const v16h*)&sH1[((y + dy) * H1W + (x + dx)) * 32 + half * 16];
                acc = __builtin_amdgcn_wmma_f32_16x16x32_f16(
                        false, A2[tap], false, bb, (short)0, acc, false, false);
            }
            // channel 8 of pixel p lives in lane lr+16, VGPR0
            const float c8 = __shfl_xor(acc[0], 16, 32);
            if (half == 0) {
                float m = c8;
#pragma unroll
                for (int i = 0; i < 8; ++i) m = fmaxf(m, acc[i]);
                float e[9];
                float sum = 0.0f;
#pragma unroll
                for (int i = 0; i < 8; ++i) { e[i] = __expf(acc[i] - m); sum += e[i]; }
                e[8] = __expf(c8 - m); sum += e[8];
                const float inv = 1.0f / sum;
                float r = 0.0f;
#pragma unroll
                for (int tap = 0; tap < 9; ++tap) {
                    const int dy = tap / 3;
                    const int dx = tap % 3;
                    r += e[tap] * sDepth[(y + dy) * H1W + (x + dx)];
                }
                out[out_base + (size_t)y * W_IMG + x] = r * inv;
            }
        }
    }
}

extern "C" void kernel_launch(void* const* d_in, const int* in_sizes, int n_in,
                              void* d_out, int out_size, void* d_ws, size_t ws_size,
                              hipStream_t stream) {
    const float* depth = (const float*)d_in[0];
    const float* img   = (const float*)d_in[1];
    const float* w1    = (const float*)d_in[2];
    const float* b1    = (const float*)d_in[3];
    const float* gamma = (const float*)d_in[4];
    const float* beta  = (const float*)d_in[5];
    const float* rmean = (const float*)d_in[6];
    const float* rvar  = (const float*)d_in[7];
    const float* w2    = (const float*)d_in[8];
    float* out = (float*)d_out;

    const int B = in_sizes[0] / (H_IMG * W_IMG);
    dim3 grid(W_IMG / TW, H_IMG / TH, B);
    FastPropagation_82205674045866_kernel<<<grid, dim3(256), 0, stream>>>(
        depth, img, w1, b1, gamma, beta, rmean, rvar, w2, out);
}
